// RateMatrixOutputHead_75239237091528
// MI455X (gfx1250) — matrix operand order repaired and checked by reading, not compiled, not run on hardware
//
#include <hip/hip_runtime.h>
#include <hip/hip_bf16.h>
#include <cstdint>

// ---------------------------------------------------------------------------
// Types for CDNA5 WMMA (wave32, 16x16x32 bf16 -> f32)
// ---------------------------------------------------------------------------
typedef __attribute__((ext_vector_type(16))) __bf16 v16bf;
typedef __attribute__((ext_vector_type(8)))  __bf16 v8bf;
typedef __attribute__((ext_vector_type(8)))  float  v8f;

union bfu { v16bf v; v8bf h[2]; };

__device__ __forceinline__ __bf16 f2bf(float f) {
  // round-to-nearest-even f32 -> bf16
  union { float f; uint32_t u; } x; x.f = f;
  uint32_t r = x.u + 0x7FFFu + ((x.u >> 16) & 1u);
  uint16_t hb = (uint16_t)(r >> 16);
  return __builtin_bit_cast(__bf16, hb);
}

// CDNA5 async global->LDS copy (ASYNCcnt tracked). lds_off = byte offset in
// LDS aperture (low 32 bits of the generic shared pointer).
__device__ __forceinline__ void async_b64(uint32_t lds_off, const void* gaddr) {
  asm volatile("global_load_async_to_lds_b64 %0, %1, off"
               :: "v"(lds_off), "v"(gaddr) : "memory");
}
__device__ __forceinline__ void wait_async0() {
  asm volatile("s_wait_asynccnt 0x0" ::: "memory");
}

static constexpr int D    = 1280;   // embed dim
static constexpr int ROWS = 8192;   // B*L
static constexpr int NST  = 20;     // states
static constexpr int MTRI = 190;    // n*(n-1)/2
static constexpr int NP   = 256;    // padded projection width (20 + 190 -> 256)
static constexpr int KSTEPS = D / 32;
static constexpr int LROW = 40;     // LDS row stride in bf16 elems (80B, padded)

// ---------------------------------------------------------------------------
// Kernel 0a: hx f32 -> bf16 row-major (packed 8B stores)
// ---------------------------------------------------------------------------
__global__ void __launch_bounds__(256)
convert_hx_kernel(const float* __restrict__ hx, __bf16* __restrict__ hxb) {
  size_t i = (size_t)blockIdx.x * 256 + threadIdx.x;  // float4 index
  float4 v = ((const float4*)hx)[i];
  union { __bf16 b[4]; uint64_t u; } o;
  o.b[0] = f2bf(v.x); o.b[1] = f2bf(v.y); o.b[2] = f2bf(v.z); o.b[3] = f2bf(v.w);
  ((uint64_t*)hxb)[i] = o.u;
}

// ---------------------------------------------------------------------------
// Kernel 0b: Wt[n][k] = bf16(dense_w[k][n])  (transpose via LDS tile)
// ---------------------------------------------------------------------------
__global__ void __launch_bounds__(256)
transpose_w_kernel(const float* __restrict__ W, __bf16* __restrict__ Wt) {
  __shared__ float tile[32][33];
  const int tx = threadIdx.x & 31, ty = threadIdx.x >> 5;   // 32 x 8
#pragma unroll
  for (int i = 0; i < 4; ++i) {
    int k = blockIdx.y * 32 + ty + i * 8;
    tile[ty + i * 8][tx] = W[(size_t)k * D + blockIdx.x * 32 + tx];
  }
  __syncthreads();
#pragma unroll
  for (int i = 0; i < 4; ++i) {
    int n = blockIdx.x * 32 + ty + i * 8;
    Wt[(size_t)n * D + blockIdx.y * 32 + tx] = f2bf(tile[tx][ty + i * 8]);
  }
}

// ---------------------------------------------------------------------------
// Kernel 0c: Wct[n][k] = bf16 of combined padded head weight (transposed):
//   n<20 -> theta_w[k][n]; 20<=n<210 -> Theta_w[k][n-20]; else 0
// ---------------------------------------------------------------------------
__global__ void __launch_bounds__(256)
build_wct_kernel(const float* __restrict__ theta_w, const float* __restrict__ Theta_w,
                 __bf16* __restrict__ Wct) {
  const int n = blockIdx.x;
#pragma unroll
  for (int i = 0; i < 5; ++i) {
    int k = threadIdx.x + i * 256;
    float v = (n < NST) ? theta_w[(size_t)k * NST + n]
            : ((n < NST + MTRI) ? Theta_w[(size_t)k * MTRI + (n - NST)] : 0.0f);
    Wct[(size_t)n * D + k] = f2bf(v);
  }
}

// ---------------------------------------------------------------------------
// Kernel 1: h_pre = hxb @ Wt^T + dense_b  (bf16 WMMA, f32 accum)
// BM=128, BN=128, BK=32; 8 waves (2M x 4N); async double-buffered LDS.
// ---------------------------------------------------------------------------
__global__ void __launch_bounds__(256)
dense_gemm_kernel(const __bf16* __restrict__ hxb, const __bf16* __restrict__ Wt,
                  const float* __restrict__ bias, float* __restrict__ out) {
  __shared__ __bf16 As[2][128 * LROW];   // [row][k] rows padded to 80B; 20KB
  __shared__ __bf16 Bs[2][128 * LROW];   // [n][k]; 20KB

  const int t    = threadIdx.x;
  const int lane = t & 31;
  const int w    = t >> 5;
  const int wm   = w & 1;
  const int wn   = w >> 1;
  const int lr   = lane & 15;
  const int half = lane >> 4;
  const int mBase = blockIdx.y * 128;
  const int nBase = blockIdx.x * 128;

  const uint32_t aBase = (uint32_t)(uintptr_t)&As[0][0];
  const uint32_t bBase = (uint32_t)(uintptr_t)&Bs[0][0];
  const int      row   = t >> 3;            // 0..31? no: e-based below
  (void)row;

  v8f c[4][2] = {};

  auto issue = [&](int buf, int k0) {
#pragma unroll
    for (int i = 0; i < 4; ++i) {
      int e     = t + i * 256;              // 1024 x 8B = 8KB per matrix
      int r     = e >> 3;                   // 0..127
      int inner = (e & 7) * 8;              // byte within 64B row
      async_b64(aBase + (uint32_t)(buf * 128 * LROW * 2 + r * 80 + inner),
                (const char*)(hxb + (size_t)(mBase + r) * D + k0) + inner);
      async_b64(bBase + (uint32_t)(buf * 128 * LROW * 2 + r * 80 + inner),
                (const char*)(Wt + (size_t)(nBase + r) * D + k0) + inner);
    }
  };

  issue(0, 0);
  for (int s = 0; s < KSTEPS; ++s) {
    const int cur = s & 1;
    wait_async0();
    __syncthreads();                        // buf[cur] ready; buf[1-cur] free
    if (s + 1 < KSTEPS) issue(1 - cur, (s + 1) * 32);

    const __bf16* Ab = &As[cur][0];
    const __bf16* Bb = &Bs[cur][0];
    v16bf a[4], b[2];
#pragma unroll
    for (int mt = 0; mt < 4; ++mt) {
      const __bf16* p = Ab + (wm * 64 + mt * 16 + lr) * LROW + half * 8;
      bfu u; u.h[0] = *(const v8bf*)p; u.h[1] = *(const v8bf*)(p + 16);
      a[mt] = u.v;
    }
#pragma unroll
    for (int nt = 0; nt < 2; ++nt) {
      const __bf16* p = Bb + (wn * 32 + nt * 16 + lr) * LROW + half * 8;
      bfu u; u.h[0] = *(const v8bf*)p; u.h[1] = *(const v8bf*)(p + 16);
      b[nt] = u.v;
    }
#pragma unroll
    for (int mt = 0; mt < 4; ++mt)
#pragma unroll
      for (int nt = 0; nt < 2; ++nt)
        c[mt][nt] = __builtin_amdgcn_wmma_f32_16x16x32_bf16(
            false, a[mt], false, b[nt], (short)0, c[mt][nt], false, false);
  }

  // epilogue: +bias, store f32 pre-activation
#pragma unroll
  for (int nt = 0; nt < 2; ++nt) {
    int col = nBase + wn * 32 + nt * 16 + lr;
    float bv = bias[col];
#pragma unroll
    for (int mt = 0; mt < 4; ++mt) {
      int rbase = mBase + wm * 64 + mt * 16 + half * 8;
#pragma unroll
      for (int v = 0; v < 8; ++v)
        out[(size_t)(rbase + v) * D + col] = c[mt][nt][v] + bv;
    }
  }
}

// ---------------------------------------------------------------------------
// Kernel 2: h = LN(gelu_erf(h_pre)) -> bf16. One block per row.
// ---------------------------------------------------------------------------
__global__ void __launch_bounds__(256)
gelu_ln_kernel(const float* __restrict__ pre, const float* __restrict__ lnw,
               const float* __restrict__ lnb, __bf16* __restrict__ hout) {
  __shared__ float rs[256];
  __shared__ float rs2[256];
  const int r = blockIdx.x;
  const int t = threadIdx.x;
  const float* row = pre + (size_t)r * D;

  float g[5];
  float s = 0.f, s2 = 0.f;
#pragma unroll
  for (int i = 0; i < 5; ++i) {
    float x = row[t + i * 256];
    float gg = 0.5f * x * (1.0f + erff(x * 0.70710678118654752440f));
    g[i] = gg; s += gg; s2 += gg * gg;
  }
  rs[t] = s; rs2[t] = s2;
  __syncthreads();
  for (int st = 128; st > 0; st >>= 1) {
    if (t < st) { rs[t] += rs[t + st]; rs2[t] += rs2[t + st]; }
    __syncthreads();
  }
  const float mu  = rs[0] * (1.0f / D);
  const float var = rs2[0] * (1.0f / D) - mu * mu;
  const float inv = rsqrtf(var + 1e-5f);
#pragma unroll
  for (int i = 0; i < 5; ++i) {
    int cidx = t + i * 256;
    float y = (g[i] - mu) * inv * lnw[cidx] + lnb[cidx];
    hout[(size_t)r * D + cidx] = f2bf(y);
  }
}

// ---------------------------------------------------------------------------
// Kernel 3: fused head. One block per 16 rows.
//   P[16,256] = h[16,1280] @ Wct^T + bias  (WMMA, async dbl-buffered B)
//   pi = softmax(P[:, :20]); softplus(P[:, 20:210]) -> symmetric scatter;
//   Q = S * sqrt(pi_j)/sqrt(pi_i);  Q_ii = -row_sum.
// ---------------------------------------------------------------------------
__global__ void __launch_bounds__(256)
head_kernel(const __bf16* __restrict__ h, const __bf16* __restrict__ Wct,
            const float* __restrict__ theta_b, const float* __restrict__ Theta_b,
            float* __restrict__ out) {
  __shared__ __bf16 Bs[2][NP * LROW];   // [n][k] padded rows; 40KB total
  __shared__ float  Ps[16 * NP];        // 16KB
  __shared__ float  bias_s[NP];
  __shared__ float  sqpi[16 * NST];
  __shared__ float  Qs[16 * NST * NST]; // 25.6KB

  const int t     = threadIdx.x;
  const int lane  = t & 31;
  const int w     = t >> 5;
  const int lr    = lane & 15;
  const int half  = lane >> 4;
  const int mBase = blockIdx.x * 16;

  bias_s[t] = (t < NST) ? theta_b[t]
            : ((t < NST + MTRI) ? Theta_b[t - NST] : 0.0f);

  const uint32_t bBase = (uint32_t)(uintptr_t)&Bs[0][0];
  auto issueB = [&](int buf, int k0) {
#pragma unroll
    for (int i = 0; i < 8; ++i) {
      int e     = t + i * 256;            // 2048 x 8B = 16KB
      int n     = e >> 3;                 // 0..255
      int inner = (e & 7) * 8;
      async_b64(bBase + (uint32_t)(buf * NP * LROW * 2 + n * 80 + inner),
                (const char*)(Wct + (size_t)n * D + k0) + inner);
    }
  };

  v8f c[2] = {};
  issueB(0, 0);
  for (int s = 0; s < KSTEPS; ++s) {
    const int cur = s & 1;
    const int k0  = s * 32;
    wait_async0();
    __syncthreads();
    if (s + 1 < KSTEPS) issueB(1 - cur, k0 + 32);

    // A fragment straight from global bf16 h (ISA bf16 A layout)
    const __bf16* ap = h + (size_t)(mBase + lr) * D + k0 + half * 8;
    bfu ua; ua.h[0] = *(const v8bf*)ap; ua.h[1] = *(const v8bf*)(ap + 16);

    const __bf16* Bb = &Bs[cur][0];
#pragma unroll
    for (int nt = 0; nt < 2; ++nt) {
      const __bf16* bp = Bb + (w * 32 + nt * 16 + lr) * LROW + half * 8;
      bfu ub; ub.h[0] = *(const v8bf*)bp; ub.h[1] = *(const v8bf*)(bp + 16);
      c[nt] = __builtin_amdgcn_wmma_f32_16x16x32_bf16(
          false, ua.v, false, ub.v, (short)0, c[nt], false, false);
    }
  }

  // ---- P -> LDS (+bias) ----
#pragma unroll
  for (int nt = 0; nt < 2; ++nt) {
    int n = w * 32 + nt * 16 + lr;
#pragma unroll
    for (int v = 0; v < 8; ++v) {
      int r = half * 8 + v;
      Ps[r * NP + n] = c[nt][v] + bias_s[n];
    }
  }
  __syncthreads();

  // ---- phase 1: softmax over 20 states, keep sqrt(pi) ----
  if (t < 16) {
    const int r = t;
    float mx = -1e30f;
    for (int i = 0; i < NST; ++i) mx = fmaxf(mx, Ps[r * NP + i]);
    float e[NST], sum = 0.f;
    for (int i = 0; i < NST; ++i) { e[i] = expf(Ps[r * NP + i] - mx); sum += e[i]; }
    const float inv = 1.0f / sum;
    for (int i = 0; i < NST; ++i) sqpi[r * NST + i] = sqrtf(e[i] * inv);
  }
  __syncthreads();

  // ---- phase 2: off-diagonal Q entries ----
  for (int e = t; e < 16 * NST * NST; e += 256) {
    int r   = e / (NST * NST);
    int idx = e - r * (NST * NST);
    int i = idx / NST, j = idx - i * NST;
    float q = 0.0f;
    if (i != j) {
      int a = i < j ? i : j;
      int b = i < j ? j : i;
      int tri = (a * (2 * NST - 1 - a)) / 2 + (b - a - 1);
      float x  = Ps[r * NP + NST + tri];
      float sp = fmaxf(x, 0.0f) + log1pf(expf(-fabsf(x)));   // softplus
      q = sp * sqpi[r * NST + j] / sqpi[r * NST + i];
    }
    Qs[e] = q;
  }
  __syncthreads();

  // ---- phase 3: diagonal = -row sum ----
  for (int e = t; e < 16 * NST; e += 256) {
    int r = e / NST, i = e - r * NST;
    float s = 0.f;
    for (int j = 0; j < NST; ++j) s += Qs[r * NST * NST + i * NST + j];
    Qs[r * NST * NST + i * NST + i] = -s;
  }
  __syncthreads();

  // ---- coalesced store ----
  for (int e = t; e < 16 * NST * NST; e += 256)
    out[(size_t)mBase * NST * NST + e] = Qs[e];
}

// ---------------------------------------------------------------------------
// Launch
// ---------------------------------------------------------------------------
extern "C" void kernel_launch(void* const* d_in, const int* in_sizes, int n_in,
                              void* d_out, int out_size, void* d_ws, size_t ws_size,
                              hipStream_t stream) {
  const float* hx      = (const float*)d_in[0];
  const float* dense_w = (const float*)d_in[1];
  const float* dense_b = (const float*)d_in[2];
  const float* ln_w    = (const float*)d_in[3];
  const float* ln_b    = (const float*)d_in[4];
  const float* theta_w = (const float*)d_in[5];
  const float* theta_b = (const float*)d_in[6];
  const float* Theta_w = (const float*)d_in[7];
  const float* Theta_b = (const float*)d_in[8];
  float* out = (float*)d_out;

  // workspace layout:
  //   [0            ) h_pre f32 : 41.94 MB
  //   [41.94 MB     ) hxb bf16  : 20.97 MB  (reused as h_bf after GEMM)
  //   [62.91 MB     ) Wt  bf16  :  3.28 MB
  //   [66.19 MB     ) Wct bf16  :  0.66 MB
  char* wsb = (char*)d_ws;
  float*  h_pre = (float*)wsb;
  __bf16* hxb   = (__bf16*)(wsb + (size_t)ROWS * D * 4);
  __bf16* h_bf  = hxb;   // reuse after dense GEMM consumed hxb
  __bf16* Wt    = (__bf16*)(wsb + (size_t)ROWS * D * 6);
  __bf16* Wct   = (__bf16*)(wsb + (size_t)ROWS * D * 6 + (size_t)D * D * 2);

  convert_hx_kernel<<<ROWS * D / 4 / 256, 256, 0, stream>>>(hx, hxb);
  transpose_w_kernel<<<dim3(D / 32, D / 32), 256, 0, stream>>>(dense_w, Wt);
  build_wct_kernel<<<NP, 256, 0, stream>>>(theta_w, Theta_w, Wct);

  dim3 g1(D / 128, ROWS / 128);
  dense_gemm_kernel<<<g1, 256, 0, stream>>>(hxb, Wt, dense_b, h_pre);
  gelu_ln_kernel<<<ROWS, 256, 0, stream>>>(h_pre, ln_w, ln_b, h_bf);
  head_kernel<<<ROWS / 16, 256, 0, stream>>>(h_bf, Wct, theta_b, Theta_b, out);
}
